// GNN_LinearAttn_87445534146903
// MI455X (gfx1250) — matrix operand order repaired and checked by reading, not compiled, not run on hardware
//
#include <hip/hip_runtime.h>
#include <hip/hip_bf16.h>
#include <math.h>

// Problem constants (reference: B=8, N=2048, D=512)
constexpr int Bb = 8;
constexpr int Nn = 2048;
constexpr int Dd = 512;
constexpr float SCALE = 0.04419417382415922f; // 1/sqrt(512)

typedef __bf16 v16bf __attribute__((ext_vector_type(16)));
typedef float  v8f   __attribute__((ext_vector_type(8)));
typedef float  vf4   __attribute__((ext_vector_type(4)));
typedef unsigned int u32x4 __attribute__((ext_vector_type(4)));
typedef unsigned int u32x8 __attribute__((ext_vector_type(8)));

// ---------------------------------------------------------------------------
// WMMA helpers (CDNA5 wave32, D = A x B + C, 16x16x32 bf16 -> f32 acc)
// ---------------------------------------------------------------------------
static __device__ __forceinline__ v8f wmma_bf16(v16bf a, v16bf b, v8f c) {
  // (neg_a, A, neg_b, B, c_mod, C, reuse_a, reuse_b)
  return __builtin_amdgcn_wmma_f32_16x16x32_bf16(false, a, false, b, (short)0, c,
                                                 false, false);
}

// A fragment 16(M)x32(K) bf16, row-major source, ld in elements.
// Layout (ISA 7.12.2): lanes 0-15 -> M=lane, lanes 16-31 -> M=lane-16;
// halves 0..7 hold K = hi*8 + 0..7, halves 8..15 hold K = hi*8 + 16..23.
static __device__ __forceinline__ v16bf frag_a_load(const __bf16* p, int ld,
                                                    int row0, int k0) {
  const int lane = threadIdx.x & 31;
  const int lo = lane & 15, hi = lane >> 4;
  const __bf16* r = p + (size_t)(row0 + lo) * ld + k0 + hi * 8;
  union { v16bf v; uint4 u[2]; } t;
  t.u[0] = *(const uint4*)(r);
  t.u[1] = *(const uint4*)(r + 16);
  return t.v;
}

// B fragment 32(K)x16(N) bf16. Source is the row-major matrix whose ROWS are
// the output columns (W for X*W^T, QK rows for QK*QK^T, xg^T rows for S*xg).
// Layout: lane -> N=lane&15, halves 0..7 hold K = hi*8 + 0..7, halves 8..15 hold K = hi*8 + 16..23.
static __device__ __forceinline__ v16bf frag_b_load(const __bf16* p, int ld,
                                                    int col0, int k0) {
  const int lane = threadIdx.x & 31;
  const int lo = lane & 15, hi = lane >> 4;
  const __bf16* r = p + (size_t)(col0 + lo) * ld + k0 + hi * 8;
  union { v16bf v; uint4 u[2]; } t;
  t.u[0] = *(const uint4*)(r);
  t.u[1] = *(const uint4*)(r + 16);
  return t.v;
}

static __device__ __forceinline__ float sigmoidf_(float z) {
  return 1.0f / (1.0f + __expf(-z));
}

// ---------------------------------------------------------------------------
// Tensor Data Mover via inline asm (builtins absent on this toolchain; the
// gfx1250 MC assembler still accepts the instruction). 2D tile load with LDS
// row padding; D# group0 (4 SGPRs) + group1 (8 SGPRs), groups 2/3 = NULL.
// ---------------------------------------------------------------------------
static __device__ __forceinline__ void tdm_load_2d(const void* gptr,
                                                   unsigned lds_off,
                                                   unsigned tensor_d0,
                                                   unsigned tensor_d1,
                                                   unsigned tile_d0,
                                                   unsigned tile_d1,
                                                   unsigned stride0,
                                                   unsigned data_size_code,
                                                   unsigned pad_en,
                                                   unsigned pad_interval_code,
                                                   unsigned pad_amount_code) {
  const unsigned long long ga = (unsigned long long)gptr;
  u32x4 g0;
  g0[0] = 1u;                                             // count=1, user mode
  g0[1] = lds_off;                                        // lds_addr (bytes)
  g0[2] = (unsigned)ga;                                   // global_addr[31:0]
  g0[3] = (unsigned)((ga >> 32) & 0x1FFFFFFu) | (2u << 30); // addr[56:32]|type=2

  u32x8 g1;
  g1[0] = (data_size_code << 16) | (pad_en << 20) |
          (pad_interval_code << 22) | (pad_amount_code << 25);
  g1[1] = (tensor_d0 & 0xFFFFu) << 16;                    // tensor_dim0 lo16
  g1[2] = (tensor_d0 >> 16) | ((tensor_d1 & 0xFFFFu) << 16);
  g1[3] = (tensor_d1 >> 16) | (tile_d0 << 16);            // tile_dim0
  g1[4] = tile_d1;                                        // tile_dim1 (dim2=0)
  g1[5] = stride0;                                        // tensor_dim0_stride lo
  g1[6] = 0;
  g1[7] = 0;

  asm volatile("tensor_load_to_lds %0, %1"
               :
               : "s"(g0), "s"(g1)
               : "memory");
}

static __device__ __forceinline__ void wait_tensorcnt0() {
  asm volatile("s_wait_tensorcnt 0x0" ::: "memory");
}

// ---------------------------------------------------------------------------
// Kernel 1: convert the three DxD weight matrices to bf16
// ---------------------------------------------------------------------------
__global__ __launch_bounds__(256) void k_cvtw(const float* __restrict__ Wqk,
                                              const float* __restrict__ Wl,
                                              const float* __restrict__ Wr,
                                              __bf16* __restrict__ wq,
                                              __bf16* __restrict__ wl,
                                              __bf16* __restrict__ wr) {
  int idx = blockIdx.x * 256 + threadIdx.x;
  if (idx < Dd * Dd) {
    wq[idx] = (__bf16)Wqk[idx];
    wl[idx] = (__bf16)Wl[idx];
    wr[idx] = (__bf16)Wr[idx];
  }
}

// ---------------------------------------------------------------------------
// Kernel 2: deg = rowsum(adj); gate = sigmoid(deg*Wd + bd); xg = x*gate
// Emits xg (bf16, [B,N,D]) and xg^T (bf16, [B,D,N]).
// adj is a 134MB single-use stream -> non-temporal loads (keep L2 for reuse).
// ---------------------------------------------------------------------------
__global__ __launch_bounds__(256) void k_gate(const float* __restrict__ x,
                                              const float* __restrict__ adj,
                                              const float* __restrict__ Wd,
                                              const float* __restrict__ bd,
                                              __bf16* __restrict__ xg,
                                              __bf16* __restrict__ xgT) {
  __shared__ float red[256];
  const int n = blockIdx.x, b = blockIdx.y, tid = threadIdx.x;
  const vf4* a4 = (const vf4*)(adj + ((size_t)b * Nn + n) * Nn);
  float s = 0.f;
  for (int i = tid; i < Nn / 4; i += 256) {
    vf4 v = __builtin_nontemporal_load(a4 + i);
    s += v[0] + v[1] + v[2] + v[3];
  }
  red[tid] = s;
  __syncthreads();
  for (int st = 128; st > 0; st >>= 1) {
    if (tid < st) red[tid] += red[tid + st];
    __syncthreads();
  }
  const float deg = red[0];
  const size_t rowbase = ((size_t)b * Nn + n) * Dd;
  for (int d = tid; d < Dd; d += 256) {
    float g = sigmoidf_(deg * Wd[d] + bd[d]);
    float v = x[rowbase + d] * g;
    __bf16 bv = (__bf16)v;
    xg[rowbase + d] = bv;
    xgT[((size_t)b * Dd + d) * Nn + n] = bv;
  }
}

// ---------------------------------------------------------------------------
// Kernel 3: QK = sigmoid(xg @ Wqk^T + bqk)  (bf16 out)
// Block: 256 thr = 8 waves; block tile 128x128, wave tile 32x64.
// ---------------------------------------------------------------------------
__global__ __launch_bounds__(256) void k_qk(const __bf16* __restrict__ xg,
                                            const __bf16* __restrict__ wq,
                                            const float* __restrict__ bqk,
                                            __bf16* __restrict__ QK) {
  const int b = blockIdx.z;
  const int wave = threadIdx.x >> 5;
  const int n0 = blockIdx.y * 128 + (wave & 3) * 32;
  const int c0 = blockIdx.x * 128 + (wave >> 2) * 64;
  const int lane = threadIdx.x & 31, lo = lane & 15, hi = lane >> 4;
  const __bf16* X = xg + (size_t)b * Nn * Dd;

  v8f acc[2][4];
  v8f z = {0.f, 0.f, 0.f, 0.f, 0.f, 0.f, 0.f, 0.f};
  for (int i = 0; i < 2; ++i)
    for (int j = 0; j < 4; ++j) acc[i][j] = z;

  for (int k0 = 0; k0 < Dd; k0 += 32) {
    v16bf a0 = frag_a_load(X, Dd, n0, k0);
    v16bf a1 = frag_a_load(X, Dd, n0 + 16, k0);
#pragma unroll
    for (int j = 0; j < 4; ++j) {
      v16bf bf = frag_b_load(wq, Dd, c0 + j * 16, k0);
      acc[0][j] = wmma_bf16(a0, bf, acc[0][j]);
      acc[1][j] = wmma_bf16(a1, bf, acc[1][j]);
    }
  }
#pragma unroll
  for (int j = 0; j < 4; ++j) {
    const int col = c0 + j * 16 + lo;
    const float bias = bqk[col];
#pragma unroll
    for (int i = 0; i < 2; ++i)
#pragma unroll
      for (int r = 0; r < 8; ++r) {
        const int row = n0 + i * 16 + r + 8 * hi;
        float v = sigmoidf_(acc[i][j][r] + bias);
        QK[((size_t)b * Nn + row) * Dd + col] = (__bf16)v;
      }
  }
}

// ---------------------------------------------------------------------------
// Kernel 4: fused masked linear attention.
//   QK row-tile (32x512 bf16) staged once into LDS via TDM (padded rows)
//   S = (QK QK^T)*adj/sqrt(D) for the 32-row tile -> LDS (bf16, 32 x 2048)
//   rowsum accumulated in deterministic per-wave LDS slots
//   O = (S @ xg) / (rowsum + 1e-6)  -> bf16
// Block: 256 thr = 8 waves; grid (N/32, B). Dynamic LDS ~162 KB (<320 KB WGP).
// ---------------------------------------------------------------------------
constexpr int SMS = 2056;                        // S row stride (elements)
constexpr int QKS_STRIDE = 520;                  // staged QK row stride (elems)
constexpr unsigned LDS_OFF_S = 1024;             // after rowsum slots
constexpr unsigned LDS_OFF_QK = LDS_OFF_S + 32u * SMS * 2u;     // 132608
constexpr unsigned LDS_TOTAL = LDS_OFF_QK + 32u * QKS_STRIDE * 2u; // 165888

__global__ __launch_bounds__(256) void k_attn(const __bf16* __restrict__ QK,
                                              const float* __restrict__ adj,
                                              const __bf16* __restrict__ xgT,
                                              __bf16* __restrict__ Obuf) {
  extern __shared__ char smem[];
  float* rowsum_p = (float*)smem;                       // [8 waves][32 rows]
  __bf16* S = (__bf16*)(smem + LDS_OFF_S);              // [32][SMS]
  __bf16* QKS = (__bf16*)(smem + LDS_OFF_QK);           // [32][QKS_STRIDE]

  const int b = blockIdx.y;
  const int n0 = blockIdx.x * 32;
  const int wave = threadIdx.x >> 5;
  const int lane = threadIdx.x & 31, lo = lane & 15, hi = lane >> 4;

  const __bf16* Q = QK + (size_t)b * Nn * Dd;
  const float* A = adj + (size_t)b * Nn * Nn;

  // ---- Stage the 32x512 QK A-tile into LDS via TDM (once per block). ----
  if (wave == 0) {
    // 2B elems, tile 512x32, pad 4 dwords (16B) after every 256 dwords (1024B)
    // -> LDS row stride 1040B = 520 elements (bank-conflict-free).
    tdm_load_2d(Q + (size_t)n0 * Dd, LDS_OFF_QK,
                /*tensor_d0=*/Dd, /*tensor_d1=*/32,
                /*tile_d0=*/Dd, /*tile_d1=*/32,
                /*stride0=*/Dd, /*data_size=*/1u,
                /*pad_en=*/1u, /*pad_interval=*/7u, /*pad_amount=*/3u);
    wait_tensorcnt0();
  }
  __syncthreads();

  // ---- Phase 1: scores. Wave w covers m in [w*256, w*256+256). ----
  float racc[2][8];
#pragma unroll
  for (int i = 0; i < 2; ++i)
#pragma unroll
    for (int r = 0; r < 8; ++r) racc[i][r] = 0.f;

  for (int t = 0; t < 8; ++t) {
    const int m0 = wave * 256 + t * 32;
    if (t < 7)  // stream-prefetch next adj tile (lane -> one of the 32 rows)
      __builtin_prefetch(A + (size_t)(n0 + lane) * Nn + m0 + 32, 0, 1);

    v8f s[2][2];
    v8f z = {0.f, 0.f, 0.f, 0.f, 0.f, 0.f, 0.f, 0.f};
    s[0][0] = z; s[0][1] = z; s[1][0] = z; s[1][1] = z;

    for (int k0 = 0; k0 < Dd; k0 += 32) {
      v16bf a0 = frag_a_load(QKS, QKS_STRIDE, 0, k0);   // LDS
      v16bf a1 = frag_a_load(QKS, QKS_STRIDE, 16, k0);  // LDS
      v16bf b0 = frag_b_load(Q, Dd, m0, k0);            // global (L2-resident)
      v16bf b1 = frag_b_load(Q, Dd, m0 + 16, k0);
      s[0][0] = wmma_bf16(a0, b0, s[0][0]);
      s[0][1] = wmma_bf16(a0, b1, s[0][1]);
      s[1][0] = wmma_bf16(a1, b0, s[1][0]);
      s[1][1] = wmma_bf16(a1, b1, s[1][1]);
    }
    // adjacency mask + scale; accumulate rowsums; spill bf16 scores to LDS
#pragma unroll
    for (int i = 0; i < 2; ++i)
#pragma unroll
      for (int j = 0; j < 2; ++j) {
        const int col = m0 + j * 16 + lo;
#pragma unroll
        for (int r = 0; r < 8; ++r) {
          const int lr = i * 16 + r + 8 * hi;           // local row 0..31
          const float av =
              __builtin_nontemporal_load(A + (size_t)(n0 + lr) * Nn + col);
          const float sv = s[i][j][r] * av * SCALE;
          racc[i][r] += sv;
          S[lr * SMS + col] = (__bf16)sv;
        }
      }
  }
  // cross-lane reduce within each 16-lane half-wave, write per-wave slot
#pragma unroll
  for (int i = 0; i < 2; ++i)
#pragma unroll
    for (int r = 0; r < 8; ++r) {
      float v = racc[i][r];
      v += __shfl_xor(v, 1, 16);
      v += __shfl_xor(v, 2, 16);
      v += __shfl_xor(v, 4, 16);
      v += __shfl_xor(v, 8, 16);
      if (lo == 0) rowsum_p[wave * 32 + (i * 16 + r + 8 * hi)] = v;
    }
  __syncthreads();

  // ---- Phase 2: O = S @ xg, wave w handles 64 output columns. ----
  const int d0 = wave * 64;
  const __bf16* XT = xgT + (size_t)b * Dd * Nn;
  v8f o[2][4];
  {
    v8f z = {0.f, 0.f, 0.f, 0.f, 0.f, 0.f, 0.f, 0.f};
    for (int i = 0; i < 2; ++i)
      for (int j = 0; j < 4; ++j) o[i][j] = z;
  }
  for (int m0 = 0; m0 < Nn; m0 += 32) {
    v16bf a0 = frag_a_load(S, SMS, 0, m0);              // LDS
    v16bf a1 = frag_a_load(S, SMS, 16, m0);
#pragma unroll
    for (int j = 0; j < 4; ++j) {
      v16bf bf = frag_b_load(XT, Nn, d0 + j * 16, m0);
      o[0][j] = wmma_bf16(a0, bf, o[0][j]);
      o[1][j] = wmma_bf16(a1, bf, o[1][j]);
    }
  }
  // normalize by rowsum (sum of the 8 per-wave slots) and store bf16
#pragma unroll
  for (int i = 0; i < 2; ++i)
#pragma unroll
    for (int r = 0; r < 8; ++r) {
      const int lr = i * 16 + r + 8 * hi;
      float rs = 0.f;
#pragma unroll
      for (int w = 0; w < 8; ++w) rs += rowsum_p[w * 32 + lr];
      const float inv = 1.0f / (rs + 1e-6f);
#pragma unroll
      for (int j = 0; j < 4; ++j) {
        const int col = d0 + j * 16 + lo;
        Obuf[((size_t)b * Nn + n0 + lr) * Dd + col] = (__bf16)(o[i][j][r] * inv);
      }
    }
}

// ---------------------------------------------------------------------------
// Kernel 5: out = relu(O @ Wl^T + bl + xg @ Wr^T), fp32 output (NT stores).
// ---------------------------------------------------------------------------
__global__ __launch_bounds__(256) void k_out(const __bf16* __restrict__ Obuf,
                                             const __bf16* __restrict__ xg,
                                             const __bf16* __restrict__ wl,
                                             const __bf16* __restrict__ wr,
                                             const float* __restrict__ bl,
                                             float* __restrict__ out) {
  const int b = blockIdx.z;
  const int wave = threadIdx.x >> 5;
  const int n0 = blockIdx.y * 128 + (wave & 3) * 32;
  const int c0 = blockIdx.x * 128 + (wave >> 2) * 64;
  const int lane = threadIdx.x & 31, lo = lane & 15, hi = lane >> 4;
  const __bf16* Ob = Obuf + (size_t)b * Nn * Dd;
  const __bf16* X = xg + (size_t)b * Nn * Dd;

  v8f acc[2][4];
  v8f z = {0.f, 0.f, 0.f, 0.f, 0.f, 0.f, 0.f, 0.f};
  for (int i = 0; i < 2; ++i)
    for (int j = 0; j < 4; ++j) acc[i][j] = z;

  for (int k0 = 0; k0 < Dd; k0 += 32) {
    v16bf ao0 = frag_a_load(Ob, Dd, n0, k0);
    v16bf ao1 = frag_a_load(Ob, Dd, n0 + 16, k0);
#pragma unroll
    for (int j = 0; j < 4; ++j) {
      v16bf bf = frag_b_load(wl, Dd, c0 + j * 16, k0);
      acc[0][j] = wmma_bf16(ao0, bf, acc[0][j]);
      acc[1][j] = wmma_bf16(ao1, bf, acc[1][j]);
    }
    v16bf ax0 = frag_a_load(X, Dd, n0, k0);
    v16bf ax1 = frag_a_load(X, Dd, n0 + 16, k0);
#pragma unroll
    for (int j = 0; j < 4; ++j) {
      v16bf bf = frag_b_load(wr, Dd, c0 + j * 16, k0);
      acc[0][j] = wmma_bf16(ax0, bf, acc[0][j]);
      acc[1][j] = wmma_bf16(ax1, bf, acc[1][j]);
    }
  }
#pragma unroll
  for (int j = 0; j < 4; ++j) {
    const int col = c0 + j * 16 + lo;
    const float bias = bl[col];
#pragma unroll
    for (int i = 0; i < 2; ++i)
#pragma unroll
      for (int r = 0; r < 8; ++r) {
        const int row = n0 + i * 16 + r + 8 * hi;
        float v = acc[i][j][r] + bias;
        v = v > 0.f ? v : 0.f;
        __builtin_nontemporal_store(v, out + ((size_t)b * Nn + row) * Dd + col);
      }
  }
}

// ---------------------------------------------------------------------------
// Host launcher. Inputs (fp32): x, adj, Wqk, bqk, Wl, bl, Wr, Wd, bd.
// Workspace needs ~66 MB.
// ---------------------------------------------------------------------------
extern "C" void kernel_launch(void* const* d_in, const int* in_sizes, int n_in,
                              void* d_out, int out_size, void* d_ws, size_t ws_size,
                              hipStream_t stream) {
  const float* x   = (const float*)d_in[0];
  const float* adj = (const float*)d_in[1];
  const float* Wqk = (const float*)d_in[2];
  const float* bqk = (const float*)d_in[3];
  const float* Wl  = (const float*)d_in[4];
  const float* bl  = (const float*)d_in[5];
  const float* Wr  = (const float*)d_in[6];
  const float* Wd  = (const float*)d_in[7];
  const float* bd  = (const float*)d_in[8];
  float* out = (float*)d_out;

  char* ws = (char*)d_ws;
  const size_t sz_act = (size_t)Bb * Nn * Dd * sizeof(__bf16); // 16 MB
  const size_t sz_w = (size_t)Dd * Dd * sizeof(__bf16);        // 512 KB
  __bf16* xg  = (__bf16*)(ws);
  __bf16* xgT = (__bf16*)(ws + sz_act);
  __bf16* QK  = (__bf16*)(ws + 2 * sz_act);
  __bf16* Ob  = (__bf16*)(ws + 3 * sz_act);
  __bf16* wq  = (__bf16*)(ws + 4 * sz_act);
  __bf16* wl  = (__bf16*)(ws + 4 * sz_act + sz_w);
  __bf16* wr  = (__bf16*)(ws + 4 * sz_act + 2 * sz_w);
  (void)ws_size; (void)in_sizes; (void)n_in; (void)out_size;

  k_cvtw<<<(Dd * Dd + 255) / 256, 256, 0, stream>>>(Wqk, Wl, Wr, wq, wl, wr);
  k_gate<<<dim3(Nn, Bb), 256, 0, stream>>>(x, adj, Wd, bd, xg, xgT);
  k_qk<<<dim3(Dd / 128, Nn / 128, Bb), 256, 0, stream>>>(xg, wq, bqk, QK);
  k_attn<<<dim3(Nn / 32, Bb), 256, LDS_TOTAL, stream>>>(QK, adj, xgT, Ob);
  k_out<<<dim3(Dd / 128, Nn / 128, Bb), 256, 0, stream>>>(Ob, xg, wl, wr, bl, out);
}
